// FPN_46729244180768
// MI455X (gfx1250) — compile-verified, matrix operand order checked by reading, not run on hardware
//
#include <hip/hip_runtime.h>
#include <stdint.h>

typedef __attribute__((ext_vector_type(16))) __bf16 v16bf;
typedef __attribute__((ext_vector_type(8)))  float  v8f;
typedef unsigned short ushort_t;

union BFrag {
    v16bf v;
    uint4 q[2];
};

__device__ __forceinline__ ushort_t f2bf(float f) {
    uint32_t u = __builtin_bit_cast(uint32_t, f);
    uint32_t r = (u + 0x7FFFu + ((u >> 16) & 1u)) >> 16;
    return (ushort_t)r;
}
__device__ __forceinline__ float bf2f(ushort_t h) {
    uint32_t u = ((uint32_t)h) << 16;
    return __builtin_bit_cast(float, u);
}

// tensor id -> (ptr, count):  0..3 = lw0..lw3, 4..7 = ow[level]
__device__ __forceinline__ void tensor_of(int t, const float* lw0, const float* lw1,
                                          const float* lw2, const float* lw3, const float* ow,
                                          const float*& p, int& n) {
    if      (t == 0) { p = lw0; n = 256 * 256;  }
    else if (t == 1) { p = lw1; n = 256 * 512;  }
    else if (t == 2) { p = lw2; n = 256 * 1024; }
    else if (t == 3) { p = lw3; n = 256 * 2048; }
    else             { p = ow + (size_t)(t - 4) * 589824; n = 589824; }
}

// ---------------------------------------------------------------------------
// Exact median via 2-pass histogram selection (full-grid parallel)
// ---------------------------------------------------------------------------
__global__ void zero_kernel(unsigned int* __restrict__ p, int n) {
    int i = blockIdx.x * blockDim.x + threadIdx.x;
    if (i < n) p[i] = 0u;
}

__global__ void hist1_kernel(const float* lw0, const float* lw1, const float* lw2,
                             const float* lw3, const float* ow,
                             unsigned int* __restrict__ hist1) {
    int t = blockIdx.y;
    const float* p; int n;
    tensor_of(t, lw0, lw1, lw2, lw3, ow, p, n);
    int i = blockIdx.x * blockDim.x + threadIdx.x;
    if (i >= n) return;
    unsigned int b = __builtin_bit_cast(unsigned int, p[i]) & 0x7FFFFFFFu;
    atomicAdd(&hist1[(size_t)t * 65536 + (b >> 15)], 1u);
}

// sel layout: u32 [8][2][2] = {bin, rank-within-bin} for which=0,1
__global__ void select1_kernel(const unsigned int* __restrict__ hist1,
                               unsigned int* __restrict__ sel) {
    int t = blockIdx.x;
    static const int ns[8] = {65536, 131072, 262144, 524288, 589824, 589824, 589824, 589824};
    int n = ns[t];
    const unsigned int* h = hist1 + (size_t)t * 65536;
    __shared__ unsigned int psum[256];
    unsigned int local = 0;
    const unsigned int* chunk = h + threadIdx.x * 256;
    for (int j = 0; j < 256; ++j) local += chunk[j];
    psum[threadIdx.x] = local;
    __syncthreads();
    if (threadIdx.x == 0) {
        for (int which = 0; which < 2; ++which) {
            unsigned int k = (unsigned)(n / 2 - 1 + which);
            unsigned int run = 0; int c = 0;
            while (c < 256 && run + psum[c] <= k) { run += psum[c]; ++c; }
            unsigned int bin = 65535u, rank = 0u;
            for (int j = 0; j < 256; ++j) {
                unsigned int hv = h[c * 256 + j];
                if (run + hv > k) { bin = (unsigned)(c * 256 + j); rank = k - run; break; }
                run += hv;
            }
            sel[(t * 2 + which) * 2 + 0] = bin;
            sel[(t * 2 + which) * 2 + 1] = rank;
        }
    }
}

__global__ void hist2_kernel(const float* lw0, const float* lw1, const float* lw2,
                             const float* lw3, const float* ow,
                             const unsigned int* __restrict__ sel,
                             unsigned int* __restrict__ hist2) {
    int t = blockIdx.y;
    const float* p; int n;
    tensor_of(t, lw0, lw1, lw2, lw3, ow, p, n);
    int i = blockIdx.x * blockDim.x + threadIdx.x;
    if (i >= n) return;
    unsigned int b  = __builtin_bit_cast(unsigned int, p[i]) & 0x7FFFFFFFu;
    unsigned int hi = b >> 15, lo = b & 0x7FFFu;
    for (int which = 0; which < 2; ++which)
        if (hi == sel[(t * 2 + which) * 2 + 0])
            atomicAdd(&hist2[(size_t)(t * 2 + which) * 32768 + lo], 1u);
}

__global__ void select2_kernel(const unsigned int* __restrict__ hist2,
                               const unsigned int* __restrict__ sel,
                               float* __restrict__ scales) {
    int t = blockIdx.x;
    __shared__ unsigned int psum[128];
    __shared__ float mids[2];
    for (int which = 0; which < 2; ++which) {
        const unsigned int* h = hist2 + (size_t)(t * 2 + which) * 32768;
        if (threadIdx.x < 128) {
            unsigned int local = 0;
            const unsigned int* chunk = h + threadIdx.x * 256;
            for (int j = 0; j < 256; ++j) local += chunk[j];
            psum[threadIdx.x] = local;
        }
        __syncthreads();
        if (threadIdx.x == 0) {
            unsigned int bin = sel[(t * 2 + which) * 2 + 0];
            unsigned int k   = sel[(t * 2 + which) * 2 + 1];
            unsigned int run = 0; int c = 0;
            while (c < 128 && run + psum[c] <= k) { run += psum[c]; ++c; }
            unsigned int low = 0;
            for (int j = 0; j < 256; ++j) {
                unsigned int hv = h[c * 256 + j];
                if (run + hv > k) { low = (unsigned)(c * 256 + j); break; }
                run += hv;
            }
            unsigned int bits = (bin << 15) | low;
            mids[which] = __builtin_bit_cast(float, bits);
        }
        __syncthreads();
    }
    if (threadIdx.x == 0) scales[t] = 0.5f * (mids[0] + mids[1]);
}

// ---------------------------------------------------------------------------
// Fold ternary scale into BN affine.  aff layout: [kind(4)][level(4)][256]
// ---------------------------------------------------------------------------
__global__ void affine_kernel(const float* lg, const float* lb, const float* lm, const float* lv,
                              const float* og, const float* ob, const float* om, const float* ov,
                              const float* scales, float* aff) {
    int c = threadIdx.x;
    for (int l = 0; l < 4; ++l) {
        int i = l * 256 + c;
        float sl = lg[i] * rsqrtf(lv[i] + 1e-5f);
        aff[(0 * 4 + l) * 256 + c] = scales[l] * sl;
        aff[(1 * 4 + l) * 256 + c] = lb[i] - lm[i] * sl;
        float so = og[i] * rsqrtf(ov[i] + 1e-5f);
        aff[(2 * 4 + l) * 256 + c] = scales[4 + l] * so;
        aff[(3 * 4 + l) * 256 + c] = ob[i] - om[i] * so;
    }
}

// ---------------------------------------------------------------------------
// Quantize weights to bf16 ternary {-1,0,+1}
// ---------------------------------------------------------------------------
__global__ void quant_lat_kernel(const float* __restrict__ w, int n, const float* scales, int sidx,
                                 ushort_t* __restrict__ dst) {
    int i = blockIdx.x * blockDim.x + threadIdx.x;
    if (i >= n) return;
    float s = scales[sidx];
    float q = rintf(w[i] / (s + 1e-8f));
    q = fminf(1.0f, fmaxf(-1.0f, q));
    dst[i] = f2bf(q);
}

// ow: [4][256][256][3][3] OIHW  ->  Wo: [level][tap][co][ci]
__global__ void quant_out_kernel(const float* __restrict__ ow, const float* scales,
                                 ushort_t* __restrict__ dst) {
    int i = blockIdx.x * blockDim.x + threadIdx.x;
    if (i >= 4 * 589824) return;
    int l   = i / 589824;
    int r   = i % 589824;
    int co  = r / 2304;
    int r2  = r % 2304;
    int ci  = r2 / 9;
    int tap = r2 % 9;
    float s = scales[4 + l];
    float q = rintf(ow[i] / (s + 1e-8f));
    q = fminf(1.0f, fmaxf(-1.0f, q));
    dst[((size_t)(l * 9 + tap) * 256 + co) * 256 + ci] = f2bf(q);
}

// ---------------------------------------------------------------------------
// GEMM kernels: WG = 256 threads (8 waves), tile M=256 x N=128, K step 32.
// Double-buffered LDS B tile, fragment order [n][k], 80B rows (conflict-free
// ds_load_b128).  Per-iteration order: A loads first (cheap loadcnt wait),
// next-slice staging loads (parked in VGPRs), 16 WMMAs, staging stores last
// (keeps DScnt waits on B fragments free of store noise).
// ---------------------------------------------------------------------------
#define LDS_ROW 40

__global__ void __launch_bounds__(256)
lateral_kernel(const float* __restrict__ x, int Cin, int logW, int logHW,
               const ushort_t* __restrict__ Wq,       // [256][Cin] bf16 ternary
               const float* __restrict__ alpha, const float* __restrict__ beta,
               ushort_t* __restrict__ out,            // [B][256][H][W] bf16
               const ushort_t* __restrict__ up)       // [B][256][H/2][W/2] bf16 or null
{
    const int HW   = 1 << logHW;
    const int W    = 1 << logW;
    const int n0   = blockIdx.x << 7;
    const int b    = n0 >> logHW;
    const int rem0 = n0 & (HW - 1);

    __shared__ __align__(16) ushort_t Bs[2][128][LDS_ROW];

    const int lane = threadIdx.x & 31;
    const int wid  = threadIdx.x >> 5;

    v8f acc[2][8];
    v8f zero = {0.f, 0.f, 0.f, 0.f, 0.f, 0.f, 0.f, 0.f};
#pragma unroll
    for (int m = 0; m < 2; ++m)
#pragma unroll
        for (int n = 0; n < 8; ++n) acc[m][n] = zero;

    const int stage_n   = threadIdx.x & 127;
    const int stage_k16 = (threadIdx.x >> 7) << 4;   // 0 or 16
    const float* stage_base = x + ((size_t)(b * Cin + stage_k16) << logHW) + rem0 + stage_n;

    auto stage_load = [&](int ks, float* tv) {
        const float* src = stage_base + ((size_t)(ks << 5) << logHW);
#pragma unroll
        for (int e = 0; e < 16; ++e) tv[e] = __builtin_nontemporal_load(&src[(size_t)e << logHW]);
    };
    auto stage_store = [&](int ks, const float* tv) {
        ushort_t* d = &Bs[ks & 1][stage_n][stage_k16];
#pragma unroll
        for (int e = 0; e < 16; ++e) d[e] = f2bf(tv[e]);
    };

    const int r0  = (wid * 2) * 16 + (lane & 15);
    const int kbl = (lane >> 4) << 3;
    const int nl  = lane & 15;

    const int nKsteps = Cin >> 5;
    {
        float tv[16];
        stage_load(0, tv);
        stage_store(0, tv);
    }
    for (int ks = 0; ks < nKsteps; ++ks) {
        __syncthreads();

        // A fragments first: their wait tolerates the staging burst behind them
        const int k0 = ks << 5;
        BFrag a0, a1;
        {
            const ushort_t* w0 = Wq + (size_t)r0 * Cin + k0 + kbl;
            a0.q[0] = *(const uint4*)(w0);
            a0.q[1] = *(const uint4*)(w0 + 16);
            const ushort_t* w1 = w0 + (size_t)16 * Cin;
            a1.q[0] = *(const uint4*)(w1);
            a1.q[1] = *(const uint4*)(w1 + 16);
        }

        const bool has_next = (ks + 1 < nKsteps);
        float tv[16];
        if (has_next) stage_load(ks + 1, tv);

#pragma unroll
        for (int nt = 0; nt < 8; ++nt) {
            BFrag bfr;
            bfr.q[0] = *(const uint4*)&Bs[ks & 1][nt * 16 + nl][kbl];
            bfr.q[1] = *(const uint4*)&Bs[ks & 1][nt * 16 + nl][kbl + 16];
            acc[0][nt] = __builtin_amdgcn_wmma_f32_16x16x32_bf16(false, a0.v, false, bfr.v,
                                                                 (short)0, acc[0][nt], false, false);
            acc[1][nt] = __builtin_amdgcn_wmma_f32_16x16x32_bf16(false, a1.v, false, bfr.v,
                                                                 (short)0, acc[1][nt], false, false);
        }

        if (has_next) stage_store(ks + 1, tv);
    }

    // epilogue: BN affine (+ nearest-2x upsample add) -> bf16
    const int logUpHW = logHW - 2;
    const int logUpW  = logW - 1;
#pragma unroll
    for (int mt = 0; mt < 2; ++mt) {
        const int row0 = (wid * 2 + mt) * 16 + ((lane >> 4) << 3);
#pragma unroll
        for (int nt = 0; nt < 8; ++nt) {
            const int coln = rem0 + nt * 16 + nl;
            const int upc  = ((coln >> logW) >> 1 << logUpW) + ((coln & (W - 1)) >> 1);
            ushort_t* po       = out + ((size_t)(b * 256 + row0) << logHW) + coln;
            const ushort_t* pu = up ? (up + ((size_t)(b * 256 + row0) << logUpHW) + upc) : nullptr;
#pragma unroll
            for (int j = 0; j < 8; ++j) {
                int row = row0 + j;
                float v = acc[mt][nt][j] * alpha[row] + beta[row];
                if (up) v += bf2f(pu[(size_t)j << logUpHW]);
                po[(size_t)j << logHW] = f2bf(v);
            }
        }
    }
}

__global__ void __launch_bounds__(256)
conv3_kernel(const ushort_t* __restrict__ lat, int logW, int logHW,
             const ushort_t* __restrict__ Wq9,        // [9][256][256] bf16 ternary
             const float* __restrict__ alpha, const float* __restrict__ beta,
             float* __restrict__ out)                 // [B][256][H][W] f32
{
    const int HW   = 1 << logHW;
    const int W    = 1 << logW;               // H == W (square levels)
    const int n0   = blockIdx.x << 7;
    const int b    = n0 >> logHW;
    const int rem0 = n0 & (HW - 1);

    __shared__ __align__(16) ushort_t Bs[2][128][LDS_ROW];

    const int lane = threadIdx.x & 31;
    const int wid  = threadIdx.x >> 5;

    v8f acc[2][8];
    v8f zero = {0.f, 0.f, 0.f, 0.f, 0.f, 0.f, 0.f, 0.f};
#pragma unroll
    for (int m = 0; m < 2; ++m)
#pragma unroll
        for (int n = 0; n < 8; ++n) acc[m][n] = zero;

    const int stage_n   = threadIdx.x & 127;
    const int stage_k16 = (threadIdx.x >> 7) << 4;
    const int rem_n = rem0 + stage_n;
    const int sh = rem_n >> logW, sw = rem_n & (W - 1);

    // flattened pipeline over (tap, kstep): idx = tap*8 + ks, 72 total
    auto stage_load = [&](int idx, ushort_t* tv) {
        const int tap = idx >> 3, ks = idx & 7;
        const int dh = tap / 3 - 1, dw = tap % 3 - 1;
        const int hs = sh + dh, wsx = sw + dw;
        const bool ok = (hs >= 0) && (hs < W) && (wsx >= 0) && (wsx < W);
        const size_t pixoff = ok ? (size_t)((hs << logW) + wsx) : 0;
        const ushort_t* src = lat + ((size_t)(b * 256 + (ks << 5) + stage_k16) << logHW) + pixoff;
#pragma unroll
        for (int e = 0; e < 16; ++e) tv[e] = ok ? src[(size_t)e << logHW] : (ushort_t)0;
    };
    auto stage_store = [&](int idx, const ushort_t* tv) {
        ushort_t* d = &Bs[idx & 1][stage_n][stage_k16];
#pragma unroll
        for (int e = 0; e < 16; ++e) d[e] = tv[e];
    };

    const int r0  = (wid * 2) * 16 + (lane & 15);
    const int kbl = (lane >> 4) << 3;
    const int nl  = lane & 15;

    {
        ushort_t tv[16];
        stage_load(0, tv);
        stage_store(0, tv);
    }
    for (int idx = 0; idx < 72; ++idx) {
        __syncthreads();

        const int tap = idx >> 3, k0 = (idx & 7) << 5;
        BFrag a0, a1;
        {
            const ushort_t* w0 = Wq9 + (size_t)tap * 65536 + (size_t)r0 * 256 + k0 + kbl;
            a0.q[0] = *(const uint4*)(w0);
            a0.q[1] = *(const uint4*)(w0 + 16);
            const ushort_t* w1 = w0 + (size_t)16 * 256;
            a1.q[0] = *(const uint4*)(w1);
            a1.q[1] = *(const uint4*)(w1 + 16);
        }

        const bool has_next = (idx + 1 < 72);
        ushort_t tv[16];
        if (has_next) stage_load(idx + 1, tv);

#pragma unroll
        for (int nt = 0; nt < 8; ++nt) {
            BFrag bfr;
            bfr.q[0] = *(const uint4*)&Bs[idx & 1][nt * 16 + nl][kbl];
            bfr.q[1] = *(const uint4*)&Bs[idx & 1][nt * 16 + nl][kbl + 16];
            acc[0][nt] = __builtin_amdgcn_wmma_f32_16x16x32_bf16(false, a0.v, false, bfr.v,
                                                                 (short)0, acc[0][nt], false, false);
            acc[1][nt] = __builtin_amdgcn_wmma_f32_16x16x32_bf16(false, a1.v, false, bfr.v,
                                                                 (short)0, acc[1][nt], false, false);
        }

        if (has_next) stage_store(idx + 1, tv);
    }

#pragma unroll
    for (int mt = 0; mt < 2; ++mt) {
        const int row0 = (wid * 2 + mt) * 16 + ((lane >> 4) << 3);
#pragma unroll
        for (int nt = 0; nt < 8; ++nt) {
            const int coln = rem0 + nt * 16 + nl;
            float* po = out + ((size_t)(b * 256 + row0) << logHW) + coln;
#pragma unroll
            for (int j = 0; j < 8; ++j) {
                int row = row0 + j;
                __builtin_nontemporal_store(acc[mt][nt][j] * alpha[row] + beta[row],
                                            &po[(size_t)j << logHW]);
            }
        }
    }
}

// ---------------------------------------------------------------------------
extern "C" void kernel_launch(void* const* d_in, const int* in_sizes, int n_in,
                              void* d_out, int out_size, void* d_ws, size_t ws_size,
                              hipStream_t stream) {
    // setup_inputs dict order: f0,lw0,f1,lw1,f2,lw2,f3,lw3,ow,lbn_g..lbn_v,obn_g..obn_v
    const float* f[4]  = {(const float*)d_in[0], (const float*)d_in[2],
                          (const float*)d_in[4], (const float*)d_in[6]};
    const float* lw[4] = {(const float*)d_in[1], (const float*)d_in[3],
                          (const float*)d_in[5], (const float*)d_in[7]};
    const float* ow    = (const float*)d_in[8];
    const float* lbn_g = (const float*)d_in[9];
    const float* lbn_b = (const float*)d_in[10];
    const float* lbn_m = (const float*)d_in[11];
    const float* lbn_v = (const float*)d_in[12];
    const float* obn_g = (const float*)d_in[13];
    const float* obn_b = (const float*)d_in[14];
    const float* obn_m = (const float*)d_in[15];
    const float* obn_v = (const float*)d_in[16];

    char* ws = (char*)d_ws;
    float*        scales = (float*)ws;                       // 8 f
    unsigned int* sel    = (unsigned int*)(ws + 64);         // [8][2][2] u32
    float*        aff    = (float*)(ws + 256);               // [4][4][256]
    unsigned int* hist1  = (unsigned int*)(ws + 16640);      // [8][65536]
    unsigned int* hist2  = (unsigned int*)(ws + 2113792);    // [8][2][32768]
    ushort_t*     Wl     = (ushort_t*)(ws + 4210944);        // lateral bf16
    ushort_t*     Wo     = (ushort_t*)(ws + 6177024);        // [4][9][256][256] bf16
    ushort_t*     lat    = (ushort_t*)(ws + 10895616);       // bf16 lat buffers

    static const int    logWl[4]  = {7, 6, 5, 4};            // W = 128,64,32,16
    static const int    logHWl[4] = {14, 12, 10, 8};
    static const int    Cin[4]    = {256, 512, 1024, 2048};
    static const size_t wlOff[4]  = {0, 65536, 196608, 458752};
    static const size_t latOff[4] = {0, 33554432, 41943040, 44040192};

    // exact median(|w|) via two-pass histogram selection
    zero_kernel<<<(1048576 + 255) / 256, 256, 0, stream>>>(hist1, 1048576); // hist1+hist2 (4MB)
    dim3 hg((589824 + 255) / 256, 8);
    hist1_kernel<<<hg, 256, 0, stream>>>(lw[0], lw[1], lw[2], lw[3], ow, hist1);
    select1_kernel<<<8, 256, 0, stream>>>(hist1, sel);
    hist2_kernel<<<hg, 256, 0, stream>>>(lw[0], lw[1], lw[2], lw[3], ow, sel, hist2);
    select2_kernel<<<8, 256, 0, stream>>>(hist2, sel, scales);

    affine_kernel<<<1, 256, 0, stream>>>(lbn_g, lbn_b, lbn_m, lbn_v,
                                         obn_g, obn_b, obn_m, obn_v, scales, aff);
    for (int l = 0; l < 4; ++l) {
        int n = 256 * Cin[l];
        quant_lat_kernel<<<(n + 255) / 256, 256, 0, stream>>>(lw[l], n, scales, l, Wl + wlOff[l]);
    }
    quant_out_kernel<<<(4 * 589824 + 255) / 256, 256, 0, stream>>>(ow, scales, Wo);

    // top-down: lat3 -> lat0, each fusing upsample-add of the coarser level
    for (int l = 3; l >= 0; --l) {
        int npix = 8 << logHWl[l];
        lateral_kernel<<<npix / 128, 256, 0, stream>>>(
            f[l], Cin[l], logWl[l], logHWl[l], Wl + wlOff[l],
            aff + (0 * 4 + l) * 256, aff + (1 * 4 + l) * 256,
            lat + latOff[l],
            (l < 3) ? (lat + latOff[l + 1]) : (const ushort_t*)nullptr);
    }
    // output 3x3 convs -> d_out (f32, concatenated in level order)
    for (int l = 0; l < 4; ++l) {
        int npix = 8 << logHWl[l];
        conv3_kernel<<<npix / 128, 256, 0, stream>>>(
            lat + latOff[l], logWl[l], logHWl[l], Wo + (size_t)l * 9 * 65536,
            aff + (2 * 4 + l) * 256, aff + (3 * 4 + l) * 256,
            (float*)d_out + latOff[l]);
    }
}
